// SelfAttention_57853209477605
// MI455X (gfx1250) — matrix operand order repaired and checked by reading, not compiled, not run on hardware
//
#include <hip/hip_runtime.h>
#include <hip/hip_bf16.h>
#include <stdint.h>

// Problem constants (fixed by the reference harness)
#define BB     2
#define SS     2048
#define DD     1024
#define PP     2048
#define TT     4096            // P + S
#define NHEADS 16
#define HDIM   64
#define MM     (BB*SS)         // 4096 rows of x / attention output
#define N3     (3*DD)          // 3072

typedef __attribute__((ext_vector_type(16))) __bf16 v16bf;
typedef __attribute__((ext_vector_type(8)))  float  v8f;

union Frag16 { v16bf v; uint32_t u[8]; };

__device__ __forceinline__ unsigned short f2bf(float f) {
    unsigned int u = __float_as_uint(f);
    unsigned int r = (u + 0x7FFFu + ((u >> 16) & 1u)) >> 16;   // RNE
    return (unsigned short)r;
}

// ---------------------------------------------------------------------------
// Format-conversion kernels
// ---------------------------------------------------------------------------
__global__ void cvt_x_kernel(const float* __restrict__ x, unsigned short* __restrict__ xb) {
    int i = blockIdx.x * 256 + threadIdx.x;
    xb[i] = f2bf(x[i]);
}

// W[K][N] (row-major) -> WT[N][K] bf16
__global__ void transpose_w_kernel(const float* __restrict__ W, unsigned short* __restrict__ WT,
                                   int K, int N) {
    int i = blockIdx.x * 256 + threadIdx.x;
    int k = i / N, n = i % N;
    WT[(size_t)n * K + k] = f2bf(W[i]);
}

// past_key [b][p][h][d] -> Kc [b][h][t=p][d] ; past_value -> Vc [b][h][d][t=p]
__global__ void cvt_past_kernel(const float* __restrict__ pk, const float* __restrict__ pv,
                                unsigned short* __restrict__ Kc, unsigned short* __restrict__ Vc) {
    int i = blockIdx.x * 256 + threadIdx.x;          // over B*P*H*hd = 2^22
    int d = i & (HDIM - 1);
    int h = (i >> 6) & (NHEADS - 1);
    int p = (i >> 10) & (PP - 1);
    int b = i >> 21;
    Kc[(((size_t)(b * NHEADS + h)) * TT + p) * HDIM + d] = f2bf(pk[i]);
    Vc[(((size_t)(b * NHEADS + h)) * HDIM + d) * TT + p] = f2bf(pv[i]);
}

// ---------------------------------------------------------------------------
// Fragment loaders (contiguous k order: element s of lane half h is k = 16h + s).
//  A (16x32 bf16): lane = M row; lanes 16-31 hold K=16..31; VGPR pairs K-adjacent.
//  B (32x16 bf16): lane = N col; lanes 16-31 hold K=16..31; pairs K-adjacent.
// ---------------------------------------------------------------------------
__device__ __forceinline__ void load_afrag(Frag16& a, const unsigned short* p, int half) {
    const uint32_t* ar = (const uint32_t*)p;
#pragma unroll
    for (int v = 0; v < 4; ++v) {
        a.u[v]     = ar[half * 8 + v];
        a.u[v + 4] = ar[half * 8 + 4 + v];
    }
}
__device__ __forceinline__ void load_bfrag(Frag16& b, const unsigned short* p) {
    const uint32_t* br = (const uint32_t*)p;
#pragma unroll
    for (int v = 0; v < 8; ++v) b.u[v] = br[v];
}

// ---------------------------------------------------------------------------
// Wave-level GEMM mainloop: per 32-wide K-slab, issue ALL fragment loads
// (18x global_load_b128) before the 8 WMMAs, so later WMMAs' operands are
// already resident and waits retire progressively. Single-buffered: live set
// (acc 64 + B 64 + A 8 + addrs) stays under the VGPR budget -> no spills.
// K loop unrolling disabled to keep it that way.
// A(MxK,row-major,bf16) x B(KxN) with B stored transposed BT[N][K] bf16.
// ---------------------------------------------------------------------------
__device__ __forceinline__ void wave_gemm_16x128(const unsigned short* __restrict__ A,
                                                 const unsigned short* __restrict__ BT,
                                                 int K, int row0, int col0, int lane,
                                                 v8f acc[8]) {
    const int m = lane & 15, half = lane >> 4, n = lane & 15;
    const unsigned short* arow = A + (size_t)(row0 + m) * K;
    const unsigned short* brow[8];
#pragma unroll
    for (int t = 0; t < 8; ++t) brow[t] = BT + (size_t)(col0 + t * 16 + n) * K + half * 16;

#pragma clang loop unroll(disable)
    for (int k0 = 0; k0 < K; k0 += 32) {
        __builtin_prefetch(arow + k0 + 32, 0, 1);    // global_prefetch next slab
        Frag16 a, bm[8];
        load_afrag(a, arow + k0, half);
#pragma unroll
        for (int t = 0; t < 8; ++t) load_bfrag(bm[t], brow[t] + k0);
#pragma unroll
        for (int t = 0; t < 8; ++t)
            acc[t] = __builtin_amdgcn_wmma_f32_16x16x32_bf16(
                false, a.v, false, bm[t].v, (short)0, acc[t], false, false);
    }
}

// ---------------------------------------------------------------------------
// QKV projection: qkv = x @ We + be, scattered into bf16 Q/K/V caches and the
// fp32 `present` output (fuses the concat with the past cache).
// Grid: (N3/128, MM/128), block 256 (8 waves, wave w owns rows 16w..16w+15).
// ---------------------------------------------------------------------------
__global__ void qkv_gemm_kernel(const unsigned short* __restrict__ xb,
                                const unsigned short* __restrict__ WeT,
                                const float* __restrict__ be,
                                unsigned short* __restrict__ Qb,
                                unsigned short* __restrict__ Kc,
                                unsigned short* __restrict__ Vc,
                                float* __restrict__ outp) {
    const int wave = threadIdx.x >> 5, lane = threadIdx.x & 31;
    const int row0 = blockIdx.y * 128 + wave * 16;
    const int col0 = blockIdx.x * 128;
    v8f acc[8];
#pragma unroll
    for (int t = 0; t < 8; ++t) acc[t] = (v8f){};
    wave_gemm_16x128(xb, WeT, DD, row0, col0, lane, acc);

    const size_t OUT0 = (size_t)BB * SS * DD;        // start of `present`
    const size_t PRES = (size_t)BB * SS * NHEADS * HDIM;
    const int n = lane & 15, half = lane >> 4;
#pragma unroll
    for (int v = 0; v < 8; ++v) {
        int r = row0 + v + half * 8;                 // C-layout row
        int b = r >> 11, s = r & (SS - 1);
#pragma unroll
        for (int t = 0; t < 8; ++t) {
            int c = col0 + t * 16 + n;
            float val = acc[t][v] + be[c];
            unsigned short bv = f2bf(val);
            int sec = c >> 10;                       // 0:q 1:k 2:v
            int cc  = c & (DD - 1);
            int h = cc >> 6, d = cc & (HDIM - 1);
            if (sec == 0) {
                Qb[(((size_t)(b * NHEADS + h)) * SS + s) * HDIM + d] = bv;
            } else if (sec == 1) {
                Kc[(((size_t)(b * NHEADS + h)) * TT + PP + s) * HDIM + d] = bv;
                outp[OUT0 + (((size_t)(b * SS + s)) * NHEADS + h) * HDIM + d] = val;
            } else {
                Vc[(((size_t)(b * NHEADS + h)) * HDIM + d) * TT + PP + s] = bv;
                outp[OUT0 + PRES + (((size_t)(b * SS + s)) * NHEADS + h) * HDIM + d] = val;
            }
        }
    }
}

// ---------------------------------------------------------------------------
// Flash-style attention. One wave per (b,h,16-query tile); 8 waves / block.
// Online softmax over T=4096 keys in 32-key chunks; all matmuls bf16 WMMA.
// K and V fragments for the chunk are loaded up-front so their loads overlap
// the score WMMAs and the softmax VALU/shuffle work.
// ---------------------------------------------------------------------------
__global__ void attn_kernel(const unsigned short* __restrict__ Qb,
                            const unsigned short* __restrict__ Kc,
                            const unsigned short* __restrict__ Vc,
                            const int* __restrict__ mask,
                            unsigned short* __restrict__ Ab) {
    __shared__ unsigned short plds[8][16 * 32];      // per-wave P-tile bounce buffer

    const int wave = threadIdx.x >> 5, lane = threadIdx.x & 31;
    const int gw    = blockIdx.x * 8 + wave;         // 0..4095
    const int qtile = gw & 127;                      // S/16 tiles per (b,h)
    const int bh    = gw >> 7;
    const int h = bh & (NHEADS - 1), b = bh >> 4;
    const int q0 = qtile * 16;

    const unsigned short* Qbase = Qb + (((size_t)(b * NHEADS + h)) * SS + q0) * HDIM;
    const unsigned short* Kbase = Kc + ((size_t)(b * NHEADS + h)) * TT * HDIM;
    const unsigned short* Vbase = Vc + ((size_t)(b * NHEADS + h)) * HDIM * TT;

    const int m = lane & 15, half = lane >> 4, n = lane & 15;

    // Q fragments: 16x64 = two 16x32 A-fragments, held for the whole loop.
    Frag16 qa0, qa1;
    load_afrag(qa0, Qbase + (size_t)m * HDIM, half);
    load_afrag(qa1, Qbase + (size_t)m * HDIM + 32, half);

    v8f O[4];
#pragma unroll
    for (int t = 0; t < 4; ++t) O[t] = (v8f){};
    float Mrow[8], Lrow[8];
#pragma unroll
    for (int v = 0; v < 8; ++v) { Mrow[v] = -__builtin_inff(); Lrow[v] = 0.0f; }

#pragma clang loop unroll(disable)
    for (int kb = 0; kb < TT; kb += 32) {
        // ---- issue all K and V fragment loads for this chunk up front ----
        Frag16 k00, k01, k10, k11, vb[4];
        {
            const unsigned short* kr0 = Kbase + (size_t)(kb + n) * HDIM + half * 16;
            const unsigned short* kr1 = Kbase + (size_t)(kb + 16 + n) * HDIM + half * 16;
            load_bfrag(k00, kr0);       load_bfrag(k01, kr0 + 32);   // dims 0-31 / 32-63
            load_bfrag(k10, kr1);       load_bfrag(k11, kr1 + 32);
#pragma unroll
            for (int t = 0; t < 4; ++t)
                load_bfrag(vb[t], Vbase + (size_t)(t * 16 + n) * TT + kb + half * 16);
        }

        // ---- scores: S = Q (16x64) x K^T (64x32), two 16-key N-tiles ----
        v8f c0 = (v8f){}, c1 = (v8f){};
        c0 = __builtin_amdgcn_wmma_f32_16x16x32_bf16(false, qa0.v, false, k00.v, (short)0, c0, false, false);
        c0 = __builtin_amdgcn_wmma_f32_16x16x32_bf16(false, qa1.v, false, k01.v, (short)0, c0, false, false);
        c1 = __builtin_amdgcn_wmma_f32_16x16x32_bf16(false, qa0.v, false, k10.v, (short)0, c1, false, false);
        c1 = __builtin_amdgcn_wmma_f32_16x16x32_bf16(false, qa1.v, false, k11.v, (short)0, c1, false, false);

        // ---- mask + scale + online softmax (per C-layout row = v + half*8) ----
        float corr[8];
#pragma unroll
        for (int v = 0; v < 8; ++v) {
            int row = v + half * 8;
            const int* mrow = mask + (size_t)(q0 + row) * TT + kb;
            float s0 = (mrow[n]      != 0) ? c0[v] * 0.125f : -1.25e19f;  // (-1e20)*scale
            float s1 = (mrow[16 + n] != 0) ? c1[v] * 0.125f : -1.25e19f;
            float cm = fmaxf(s0, s1);
#pragma unroll
            for (int off = 1; off < 16; off <<= 1) cm = fmaxf(cm, __shfl_xor(cm, off));
            float Mn = fmaxf(Mrow[v], cm);
            float cr = __expf(Mrow[v] - Mn);
            float p0 = __expf(s0 - Mn);
            float p1 = __expf(s1 - Mn);
            float ps = p0 + p1;
#pragma unroll
            for (int off = 1; off < 16; off <<= 1) ps += __shfl_xor(ps, off);
            Lrow[v] = Lrow[v] * cr + ps;
            Mrow[v] = Mn;
            corr[v] = cr;
            plds[wave][row * 32 + n]      = f2bf(p0);   // C-layout -> LDS 16x32 tile
            plds[wave][row * 32 + 16 + n] = f2bf(p1);
        }
#pragma unroll
        for (int t = 0; t < 4; ++t)
#pragma unroll
            for (int v = 0; v < 8; ++v) O[t][v] *= corr[v];

        // same-wave DS ops are in-order; fence the compiler + DScnt explicitly
        asm volatile("s_wait_dscnt 0" ::: "memory");

        // ---- re-read P as a 16x32 A-fragment ----
        Frag16 pf;
        {
            const uint32_t* pr = (const uint32_t*)&plds[wave][(size_t)m * 32];
#pragma unroll
            for (int v = 0; v < 4; ++v) {
                pf.u[v]     = pr[half * 8 + v];
                pf.u[v + 4] = pr[half * 8 + 4 + v];
            }
        }

        // ---- O += P (16x32) x V (32x64); V frags already resident ----
#pragma unroll
        for (int t = 0; t < 4; ++t)
            O[t] = __builtin_amdgcn_wmma_f32_16x16x32_bf16(
                false, pf.v, false, vb[t].v, (short)0, O[t], false, false);
    }

    // ---- normalize and emit bf16 rows [b*S+s][h*hd + d] for the FC GEMM ----
#pragma unroll
    for (int v = 0; v < 8; ++v) {
        int row = v + half * 8;
        float inv = 1.0f / Lrow[v];
        size_t base = ((size_t)(b * SS) + q0 + row) * DD + h * HDIM + n;
#pragma unroll
        for (int t = 0; t < 4; ++t) Ab[base + t * 16] = f2bf(O[t][v] * inv);
    }
}

// ---------------------------------------------------------------------------
// Output projection: out = Ab @ Wf + bf  (fp32 result to d_out)
// ---------------------------------------------------------------------------
__global__ void fc_gemm_kernel(const unsigned short* __restrict__ Ab,
                               const unsigned short* __restrict__ WfT,
                               const float* __restrict__ bfv,
                               float* __restrict__ outp) {
    const int wave = threadIdx.x >> 5, lane = threadIdx.x & 31;
    const int row0 = blockIdx.y * 128 + wave * 16;
    const int col0 = blockIdx.x * 128;
    v8f acc[8];
#pragma unroll
    for (int t = 0; t < 8; ++t) acc[t] = (v8f){};
    wave_gemm_16x128(Ab, WfT, DD, row0, col0, lane, acc);

    const int n = lane & 15, half = lane >> 4;
#pragma unroll
    for (int v = 0; v < 8; ++v) {
        int r = row0 + v + half * 8;
#pragma unroll
        for (int t = 0; t < 8; ++t) {
            int c = col0 + t * 16 + n;
            outp[(size_t)r * DD + c] = acc[t][v] + bfv[c];
        }
    }
}

// ---------------------------------------------------------------------------
// Host launcher
// ---------------------------------------------------------------------------
extern "C" void kernel_launch(void* const* d_in, const int* in_sizes, int n_in,
                              void* d_out, int out_size, void* d_ws, size_t ws_size,
                              hipStream_t stream) {
    (void)in_sizes; (void)n_in; (void)out_size; (void)ws_size;
    const float* x          = (const float*)d_in[0];
    const int*   mask       = (const int*)  d_in[1];
    const float* past_key   = (const float*)d_in[2];
    const float* past_value = (const float*)d_in[3];
    const float* We         = (const float*)d_in[4];
    const float* be         = (const float*)d_in[5];
    const float* Wf         = (const float*)d_in[6];
    const float* bfv        = (const float*)d_in[7];
    float* outp = (float*)d_out;

    // Workspace layout (bytes). Ab aliases xb: xb is dead after qkv_gemm and
    // kernels are stream-ordered. Total footprint: 56 MB.
    char* ws = (char*)d_ws;
    unsigned short* xb  = (unsigned short*)(ws);                    //  8 MB [4096][1024]
    unsigned short* Ab  = (unsigned short*)(ws);                    //  reuse of xb
    unsigned short* WeT = (unsigned short*)(ws + (8u  << 20));      //  6 MB [3072][1024]
    unsigned short* WfT = (unsigned short*)(ws + (14u << 20));      //  2 MB [1024][1024]
    unsigned short* Qb  = (unsigned short*)(ws + (16u << 20));      //  8 MB [b][h][s][d]
    unsigned short* Kc  = (unsigned short*)(ws + (24u << 20));      // 16 MB [b][h][t][d]
    unsigned short* Vc  = (unsigned short*)(ws + (40u << 20));      // 16 MB [b][h][d][t]

    cvt_x_kernel      <<<(MM * DD) / 256, 256, 0, stream>>>(x, xb);
    transpose_w_kernel<<<(DD * N3) / 256, 256, 0, stream>>>(We, WeT, DD, N3);
    transpose_w_kernel<<<(DD * DD) / 256, 256, 0, stream>>>(Wf, WfT, DD, DD);
    cvt_past_kernel   <<<(BB * PP * NHEADS * HDIM) / 256, 256, 0, stream>>>(past_key, past_value, Kc, Vc);

    qkv_gemm_kernel<<<dim3(N3 / 128, MM / 128), 256, 0, stream>>>(xb, WeT, be, Qb, Kc, Vc, outp);
    attn_kernel    <<<(BB * NHEADS * (SS / 16)) / 8, 256, 0, stream>>>(Qb, Kc, Vc, mask, Ab);
    fc_gemm_kernel <<<dim3(DD / 128, MM / 128), 256, 0, stream>>>(Ab, WfT, bfv, outp);
}
